// MemoryMultiHeadedAttention_6433861009893
// MI455X (gfx1250) — matrix operand order repaired and checked by reading, not compiled, not run on hardware
//
#include <hip/hip_runtime.h>
#include <hip/hip_bf16.h>
#include <math.h>

typedef __attribute__((ext_vector_type(16))) __bf16          v16bf;
typedef __attribute__((ext_vector_type(16))) unsigned short  v16us;
typedef __attribute__((ext_vector_type(8)))  float           v8f;

#define H_      8
#define DH_     64
#define DIM_    512
#define SEQ_    1024
#define MEM_    1024
#define CME_    256
#define KVLEN_  2304      // cmem + mem + x
#define BATCH_  8
#define SCALE_  0.125f    // 64^-0.5

// ---------- helpers ----------
__device__ __forceinline__ unsigned short f2bf(float f) {
  unsigned u = __builtin_bit_cast(unsigned, f);
  u += 0x7FFFu + ((u >> 16) & 1u);            // round-to-nearest-even
  return (unsigned short)(u >> 16);
}
__device__ __forceinline__ v8f wmma_bf16(v16us a, v16us b, v8f c) {
  return __builtin_amdgcn_wmma_f32_16x16x32_bf16(
      false, __builtin_bit_cast(v16bf, a),
      false, __builtin_bit_cast(v16bf, b),
      (short)0, c, false, false);
}
// A-matrix (16x32, 16-bit) per-lane K pair base for VGPR slot v (contiguous K: k = 16*half + 2*v)
__device__ __forceinline__ int a_kpair(int v, int half) {
  return 2 * v + half * 16;
}
// CDNA5 async global->LDS (ASYNCcnt-tracked DMA, bypasses VGPRs)
__device__ __forceinline__ unsigned lds_off(const void* p) {
  return (unsigned)(unsigned long long)p;     // generic LDS ptr: offset in bits [31:0]
}
__device__ __forceinline__ void async_ld_b128(unsigned dst_lds, const void* src) {
  asm volatile("global_load_async_to_lds_b128 %0, %1, off"
               :: "v"(dst_lds), "v"(src) : "memory");
}
__device__ __forceinline__ void wait_async0() {
  asm volatile("s_wait_asynccnt 0x0" ::: "memory");
}

// ---------- generic batched NT GEMM: C[M,N] = alpha * A[M,K] * B[N,K]^T (+bias) ----------
// Double-buffered LDS, tiles staged with async global->LDS DMA.
#define BM 128
#define BN 64
#define BK 32
#define LDT 40   // LDS row stride in shorts (pad keeps 16B alignment, no bank conflicts)

__global__ __launch_bounds__(256) void gemm_nt_bf16(
    const unsigned short* __restrict__ A, long lda, long sAb,
    const unsigned short* __restrict__ B, long ldb, long sBb,
    float* __restrict__ C, long ldc, long sCb,
    int K, float alpha, const float* __restrict__ bias)
{
  __shared__ unsigned short As[2][BM * LDT];
  __shared__ unsigned short Bs[2][BN * LDT];
  A += (long)blockIdx.z * sAb;
  B += (long)blockIdx.z * sBb;
  C += (long)blockIdx.z * sCb;
  const int m0 = blockIdx.y * BM, n0 = blockIdx.x * BN;
  const int tid = threadIdx.x, lane = tid & 31, wave = tid >> 5;
  const int wy = wave & 3, wx = wave >> 2;          // 4x2 wave grid, 32x32 per wave
  const int half = lane >> 4, l16 = lane & 15;

  // per-thread 16B-chunk assignment: A = 512 chunks (2/thread), B = 256 chunks (1/thread)
  const int rc = tid >> 2, qc = tid & 3;            // row 0..63, quarter 0..3 (8 shorts)
  const unsigned a_rel0 = (unsigned)((rc)      * LDT + qc * 8) * 2;
  const unsigned a_rel1 = (unsigned)((rc + 64) * LDT + qc * 8) * 2;
  const unsigned b_rel  = (unsigned)((rc)      * LDT + qc * 8) * 2;
  const long a_go0 = (long)(m0 + rc)      * lda + qc * 8;
  const long a_go1 = (long)(m0 + rc + 64) * lda + qc * 8;
  const long b_go  = (long)(n0 + rc)      * ldb + qc * 8;

  const v8f vz = (v8f)0.0f;
  v8f acc[2][2];
  acc[0][0] = vz; acc[0][1] = vz; acc[1][0] = vz; acc[1][1] = vz;

  // prologue prefetch into buffer 0
  {
    unsigned al = lds_off(&As[0][0]), bl = lds_off(&Bs[0][0]);
    async_ld_b128(al + a_rel0, A + a_go0);
    async_ld_b128(al + a_rel1, A + a_go1);
    async_ld_b128(bl + b_rel,  B + b_go);
  }
  int buf = 0;
  for (int k0 = 0; k0 < K; k0 += BK) {
    wait_async0();
    __syncthreads();                                  // buffer `buf` ready for everyone
    if (k0 + BK < K) {                                // prefetch next tile into other buffer
      unsigned al = lds_off(&As[buf ^ 1][0]), bl = lds_off(&Bs[buf ^ 1][0]);
      async_ld_b128(al + a_rel0, A + a_go0 + k0 + BK);
      async_ld_b128(al + a_rel1, A + a_go1 + k0 + BK);
      async_ld_b128(bl + b_rel,  B + b_go  + k0 + BK);
    }
    v16us af[2], bfr[2];
#pragma unroll
    for (int mi = 0; mi < 2; mi++) {
      int mr = (wy * 32 + mi * 16 + l16) * LDT;
#pragma unroll
      for (int v = 0; v < 8; v++) {
        int k = a_kpair(v, half);
        af[mi][2 * v]     = As[buf][mr + k];
        af[mi][2 * v + 1] = As[buf][mr + k + 1];
      }
    }
#pragma unroll
    for (int ni = 0; ni < 2; ni++) {   // B fragment: lane = column n, K contiguous per half
      int nr = (wx * 32 + ni * 16 + l16) * LDT;
      int kb = half * 16;
#pragma unroll
      for (int v = 0; v < 8; v++) {
        bfr[ni][2 * v]     = Bs[buf][nr + kb + 2 * v];
        bfr[ni][2 * v + 1] = Bs[buf][nr + kb + 2 * v + 1];
      }
    }
#pragma unroll
    for (int mi = 0; mi < 2; mi++)
#pragma unroll
      for (int ni = 0; ni < 2; ni++)
        acc[mi][ni] = wmma_bf16(af[mi], bfr[ni], acc[mi][ni]);
    __syncthreads();                                  // all reads of `buf` done before reuse
    buf ^= 1;
  }
#pragma unroll
  for (int mi = 0; mi < 2; mi++)
#pragma unroll
    for (int ni = 0; ni < 2; ni++) {
      int col = n0 + wx * 32 + ni * 16 + l16;
      float bv = bias ? bias[col] : 0.0f;
#pragma unroll
      for (int r = 0; r < 8; r++) {
        int row = m0 + wy * 32 + mi * 16 + r + 8 * half;   // C layout (ISA 7.12.2)
        C[(long)row * ldc + col] = acc[mi][ni][r] * alpha + bv;
      }
    }
}

// ---------- softmax row stats over attn rows (len 2304) ----------
__global__ __launch_bounds__(256) void softmax_stats(
    const float* __restrict__ attn, float2* __restrict__ stats)
{
  __shared__ float red[256];
  const long row = blockIdx.x;
  const float* p = attn + row * (long)KVLEN_;
  const int tid = threadIdx.x;
  float mx = -3.0e38f;
  for (int j = tid; j < KVLEN_; j += 256) mx = fmaxf(mx, p[j]);
  red[tid] = mx; __syncthreads();
  for (int s = 128; s > 0; s >>= 1) { if (tid < s) red[tid] = fmaxf(red[tid], red[tid + s]); __syncthreads(); }
  mx = red[0]; __syncthreads();
  float sm = 0.0f;
  for (int j = tid; j < KVLEN_; j += 256) sm += __expf(p[j] - mx);
  red[tid] = sm; __syncthreads();
  for (int s = 128; s > 0; s >>= 1) { if (tid < s) red[tid] += red[tid + s]; __syncthreads(); }
  if (tid == 0) stats[row] = make_float2(mx, 1.0f / red[0]);
}

// ---------- fused: normalize attn in place + out = attn @ v  (one wave = 16 q rows) ----------
__global__ __launch_bounds__(32) void attn_pv(
    float* __restrict__ attn, const float2* __restrict__ stats,
    const unsigned short* __restrict__ vb, float* __restrict__ outw)
{
  const int bh = blockIdx.x >> 6, it = blockIdx.x & 63, i0 = it * 16;
  const int lane = threadIdx.x, half = lane >> 4, l16 = lane & 15;
  float* S = attn + ((long)bh * SEQ_ + i0) * KVLEN_;
  const unsigned short* V = vb + (long)bh * KVLEN_ * DH_;
  const float2 st = stats[(long)bh * SEQ_ + i0 + l16]; // A-layout: each lane owns row l16
  const float mx = st.x, isum = st.y;
  const v8f vz = (v8f)0.0f;
  v8f acc[4]; acc[0] = vz; acc[1] = vz; acc[2] = vz; acc[3] = vz;
  for (int j0 = 0; j0 < KVLEN_; j0 += 32) {
    v16us pf;
#pragma unroll
    for (int v = 0; v < 8; v++) {
      int k = a_kpair(v, half);
      long idx = (long)l16 * KVLEN_ + j0 + k;
      float p0 = __expf(S[idx] - mx) * isum;
      float p1 = __expf(S[idx + 1] - mx) * isum;
      S[idx] = p0; S[idx + 1] = p1;                    // write normalized attn back
      pf[2 * v] = f2bf(p0); pf[2 * v + 1] = f2bf(p1);
    }
#pragma unroll
    for (int nt = 0; nt < 4; nt++) {                   // dh = 4 tiles of 16
      v16us vf;
      int kb = half * 16;
#pragma unroll
      for (int v = 0; v < 8; v++) {
        vf[2 * v]     = V[(long)(j0 + kb + 2 * v) * DH_ + nt * 16 + l16];
        vf[2 * v + 1] = V[(long)(j0 + kb + 2 * v + 1) * DH_ + nt * 16 + l16];
      }
      acc[nt] = wmma_bf16(pf, vf, acc[nt]);
    }
  }
  float* O = outw + ((long)bh * SEQ_ + i0) * DH_;
#pragma unroll
  for (int nt = 0; nt < 4; nt++)
#pragma unroll
    for (int r = 0; r < 8; r++)
      O[(long)(r + 8 * half) * DH_ + nt * 16 + l16] = acc[nt][r];
}

// ---------- aux loss: flash attention (j=1024 k/v vs j=256 ck/cv), MSE of outputs ----------
__global__ __launch_bounds__(32) void aux_flash(
    const unsigned short* __restrict__ qb,
    const unsigned short* __restrict__ kb,
    const unsigned short* __restrict__ vb,
    const unsigned short* __restrict__ ckb,
    const unsigned short* __restrict__ cvb,
    float* __restrict__ aux_out)
{
  __shared__ unsigned short Pt[16 * 20];
  const int bh = blockIdx.x >> 6, it = blockIdx.x & 63, i0 = it * 16;
  const int lane = threadIdx.x, half = lane >> 4, l16 = lane & 15;
  const unsigned short* Q = qb + ((long)bh * SEQ_ + i0) * DH_;
  v16us qf[2];
#pragma unroll
  for (int f = 0; f < 2; f++)
#pragma unroll
    for (int v = 0; v < 8; v++) {
      int k = a_kpair(v, half) + f * 32;
      qf[f][2 * v]     = Q[l16 * DH_ + k];
      qf[f][2 * v + 1] = Q[l16 * DH_ + k + 1];
    }
  float fa[4][8];
  float sq = 0.0f;
  const v8f vz = (v8f)0.0f;
  for (int ph = 0; ph < 2; ph++) {
    const unsigned short* Kp = ph ? ckb + (long)bh * CME_ * DH_
                                  : kb + ((long)bh * KVLEN_ + CME_) * DH_;
    const unsigned short* Vp = ph ? cvb + (long)bh * CME_ * DH_
                                  : vb + ((long)bh * KVLEN_ + CME_) * DH_;
    const int len = ph ? CME_ : MEM_;
    v8f o[4]; o[0] = vz; o[1] = vz; o[2] = vz; o[3] = vz;
    float m[8], l[8];
#pragma unroll
    for (int r = 0; r < 8; r++) { m[r] = -INFINITY; l[r] = 0.0f; }
    for (int j0 = 0; j0 < len; j0 += 16) {
      v8f s = vz;
#pragma unroll
      for (int f = 0; f < 2; f++) {                 // contraction over dh=64 (2 WMMA)
        v16us kf;
#pragma unroll
        for (int v = 0; v < 8; v++) {
          int d = half * 16 + 2 * v + f * 32;       // B frag: lane = column j
          kf[2 * v]     = Kp[(long)(j0 + l16) * DH_ + d];
          kf[2 * v + 1] = Kp[(long)(j0 + l16) * DH_ + d + 1];
        }
        s = wmma_bf16(qf[f], kf, s);
      }
      float pr[8];
#pragma unroll
      for (int r = 0; r < 8; r++) {                 // online softmax per row
        float sv = s[r] * SCALE_;
        float rmax = sv;
#pragma unroll
        for (int off = 1; off < 16; off <<= 1) rmax = fmaxf(rmax, __shfl_xor(rmax, off, 16));
        float nm = fmaxf(m[r], rmax);
        float corr = __expf(m[r] - nm);
        float p = __expf(sv - nm);
        float rsum = p;
#pragma unroll
        for (int off = 1; off < 16; off <<= 1) rsum += __shfl_xor(rsum, off, 16);
        l[r] = l[r] * corr + rsum;
        m[r] = nm;
        pr[r] = p;
#pragma unroll
        for (int nt = 0; nt < 4; nt++) o[nt][r] *= corr;
      }
      __syncthreads();                              // C-layout -> A-layout via LDS bounce
#pragma unroll
      for (int r = 0; r < 8; r++) Pt[(r + 8 * half) * 20 + l16] = f2bf(pr[r]);
      __syncthreads();
      v16us pf;
#pragma unroll
      for (int v = 0; v < 8; v++) {                 // P padded 16x16 -> 16x32 (zeros)
        int k = a_kpair(v, half);
        if (k < 16) { pf[2 * v] = Pt[l16 * 20 + k]; pf[2 * v + 1] = Pt[l16 * 20 + k + 1]; }
        else        { pf[2 * v] = 0; pf[2 * v + 1] = 0; }
      }
#pragma unroll
      for (int nt = 0; nt < 4; nt++) {
        v16us vf;
#pragma unroll
        for (int v = 0; v < 8; v++) {
          if (half == 0) {                          // only K=0..15 rows contribute
            vf[2 * v]     = Vp[(long)(j0 + 2 * v) * DH_ + nt * 16 + l16];
            vf[2 * v + 1] = Vp[(long)(j0 + 2 * v + 1) * DH_ + nt * 16 + l16];
          } else { vf[2 * v] = 0; vf[2 * v + 1] = 0; }
        }
        o[nt] = wmma_bf16(pf, vf, o[nt]);
      }
    }
#pragma unroll
    for (int nt = 0; nt < 4; nt++)
#pragma unroll
      for (int r = 0; r < 8; r++) {
        float val = o[nt][r] / l[r];
        if (ph == 0) fa[nt][r] = val;
        else { float d = val - fa[nt][r]; sq += d * d; }
      }
  }
#pragma unroll
  for (int off = 1; off < 32; off <<= 1) sq += __shfl_xor(sq, off, 32);
  if (lane == 0) atomicAdd(aux_out, sq * (1.0f / (8.0f * 8.0f * 1024.0f * 64.0f)));
}

// ---------- elementwise / repack kernels ----------
__global__ void build_kvin(const float* __restrict__ x, const float* __restrict__ mem,
                           const float* __restrict__ cmem, unsigned short* __restrict__ dst)
{
  long i = (long)blockIdx.x * 256 + threadIdx.x;
  const long total = (long)BATCH_ * KVLEN_ * DIM_;
  if (i >= total) return;
  int c = (int)(i % DIM_); long t = i / DIM_;
  int r = (int)(t % KVLEN_); int b = (int)(t / KVLEN_);
  float v;
  if (r < CME_)              v = cmem[((long)b * CME_ + r) * DIM_ + c];
  else if (r < CME_ + MEM_)  v = mem[((long)b * MEM_ + (r - CME_)) * DIM_ + c];
  else                       v = x[((long)b * SEQ_ + (r - CME_ - MEM_)) * DIM_ + c];
  dst[i] = f2bf(v);
}

__global__ void cvt_bf16(const float* __restrict__ src, unsigned short* __restrict__ dst, long n)
{
  long i = (long)blockIdx.x * 256 + threadIdx.x;
  if (i < n) dst[i] = f2bf(src[i]);
}

__global__ void repack_convw(const float* __restrict__ src, unsigned short* __restrict__ dst)
{
  long i = (long)blockIdx.x * 256 + threadIdx.x;
  if (i >= 512L * 2048) return;
  int o = (int)(i >> 11); int rem = (int)(i & 2047);
  int r = rem >> 9; int c = rem & 511;
  dst[i] = f2bf(src[((long)o * 512 + c) * 4 + r]);   // [o][i][r] -> [o][r*512+i]
}

__global__ void split_heads_cvt(const float* __restrict__ src, unsigned short* __restrict__ dst,
                                int S, int stride, int colOff)
{
  long i = (long)blockIdx.x * 256 + threadIdx.x;
  long total = (long)BATCH_ * H_ * S * 64;
  if (i >= total) return;
  int d = (int)(i & 63); long t = i >> 6;
  int s = (int)(t % S); t /= S;
  int h = (int)(t & 7); int b = (int)(t >> 3);
  dst[i] = f2bf(src[((long)b * S + s) * stride + colOff + h * 64 + d]);
}

__global__ void merge_heads_cvt(const float* __restrict__ src, unsigned short* __restrict__ dst)
{
  long i = (long)blockIdx.x * 256 + threadIdx.x;
  const long total = (long)BATCH_ * SEQ_ * DIM_;
  if (i >= total) return;
  int c = (int)(i % DIM_); long t = i / DIM_;
  int s = (int)(t % SEQ_); int b = (int)(t / SEQ_);
  int h = c >> 6, d = c & 63;
  dst[i] = f2bf(src[(((long)b * 8 + h) * SEQ_ + s) * 64 + d]);
}

__global__ void init_scalars(float* aux, float* ae)
{
  if (threadIdx.x == 0) { *aux = 0.0f; *ae = 0.0f; }
}

// ---------- host launch ----------
extern "C" void kernel_launch(void* const* d_in, const int* in_sizes, int n_in,
                              void* d_out, int out_size, void* d_ws, size_t ws_size,
                              hipStream_t stream)
{
  (void)in_sizes; (void)n_in; (void)out_size; (void)ws_size;
  const float* x     = (const float*)d_in[0];
  const float* mem   = (const float*)d_in[1];
  const float* cmem  = (const float*)d_in[2];
  const float* Wq    = (const float*)d_in[4];
  const float* Wkv   = (const float*)d_in[5];
  const float* Wout  = (const float*)d_in[6];
  const float* bout  = (const float*)d_in[7];
  const float* convw = (const float*)d_in[8];
  const float* convb = (const float*)d_in[9];
  float* out = (float*)d_out;

  // d_out layout (flat, return order)
  float* logits  = out;                 // 8*1024*512
  float* newmem  = out + 4194304;       // 8*1024*512
  float* newcmem = out + 8388608;       // 8*256*512
  float* auxp    = out + 9437184;       // 1
  float* aep     = out + 9437185;       // 1
  float* attn    = out + 9437186;       // 8*8*1024*2304

  // workspace carve-out (~202 MB), 256B aligned regions
  char* ws = (char*)d_ws;
  size_t off = 0;
  auto alloc = [&](size_t bytes) -> void* {
    void* p = ws + off;
    off = (off + bytes + 255) & ~(size_t)255;
    return p;
  };
  unsigned short* kvin   = (unsigned short*)alloc(2UL * 8 * 2304 * 512);
  unsigned short* wq_b   = (unsigned short*)alloc(2UL * 512 * 512);
  unsigned short* wkv_b  = (unsigned short*)alloc(2UL * 1024 * 512);
  unsigned short* wout_b = (unsigned short*)alloc(2UL * 512 * 512);
  unsigned short* cw_b   = (unsigned short*)alloc(2UL * 512 * 2048);
  float*          q_f    = (float*)alloc(4UL * 8 * 1024 * 512);
  float*          kv_f   = (float*)alloc(4UL * 8 * 2304 * 1024);
  unsigned short* q_b    = (unsigned short*)alloc(2UL * 8 * 8 * 1024 * 64);
  unsigned short* k_b    = (unsigned short*)alloc(2UL * 8 * 8 * 2304 * 64);
  unsigned short* v_b    = (unsigned short*)alloc(2UL * 8 * 8 * 2304 * 64);
  float2*         stats  = (float2*)alloc(8UL * 8 * 8 * 1024);
  float*          outw   = (float*)alloc(4UL * 8 * 8 * 1024 * 64);
  unsigned short* out_b  = (unsigned short*)alloc(2UL * 8 * 1024 * 512);
  unsigned short* cc_b   = (unsigned short*)alloc(2UL * 8 * 256 * 512);
  float*          ckv_f  = (float*)alloc(4UL * 8 * 256 * 1024);
  unsigned short* ck_b   = (unsigned short*)alloc(2UL * 8 * 8 * 256 * 64);
  unsigned short* cv_b   = (unsigned short*)alloc(2UL * 8 * 8 * 256 * 64);

  const dim3 blk(256);
  // bf16 staging
  build_kvin<<<(unsigned)((8L * 2304 * 512 + 255) / 256), blk, 0, stream>>>(x, mem, cmem, kvin);
  cvt_bf16<<<(unsigned)((512L * 512 + 255) / 256), blk, 0, stream>>>(Wq, wq_b, 512L * 512);
  cvt_bf16<<<(unsigned)((1024L * 512 + 255) / 256), blk, 0, stream>>>(Wkv, wkv_b, 1024L * 512);
  cvt_bf16<<<(unsigned)((512L * 512 + 255) / 256), blk, 0, stream>>>(Wout, wout_b, 512L * 512);
  repack_convw<<<(unsigned)((512L * 2048 + 255) / 256), blk, 0, stream>>>(convw, cw_b);

  // q = x @ Wq^T (x reused from kvin rows 1280..2303; batched over b)
  gemm_nt_bf16<<<dim3(512 / BN, 1024 / BM, 8), 256, 0, stream>>>(
      kvin + 1280L * 512, 512, 2304L * 512, wq_b, 512, 0,
      q_f, 512, 1024L * 512, 512, 1.0f, nullptr);
  // kv = kvin @ Wkv^T (flat M = 18432)
  gemm_nt_bf16<<<dim3(1024 / BN, 18432 / BM, 1), 256, 0, stream>>>(
      kvin, 512, 0, wkv_b, 512, 0, kv_f, 1024, 0, 512, 1.0f, nullptr);

  // head split + bf16
  split_heads_cvt<<<(unsigned)((8L * 8 * 1024 * 64 + 255) / 256), blk, 0, stream>>>(q_f, q_b, 1024, 512, 0);
  split_heads_cvt<<<(unsigned)((8L * 8 * 2304 * 64 + 255) / 256), blk, 0, stream>>>(kv_f, k_b, 2304, 1024, 0);
  split_heads_cvt<<<(unsigned)((8L * 8 * 2304 * 64 + 255) / 256), blk, 0, stream>>>(kv_f, v_b, 2304, 1024, 512);

  // dots = scale * q @ k^T -> raw scores straight into d_out attn (batched over 64 b*h)
  gemm_nt_bf16<<<dim3(2304 / BN, 1024 / BM, 64), 256, 0, stream>>>(
      q_b, 64, 1024L * 64, k_b, 64, 2304L * 64,
      attn, 2304, 1024L * 2304, 64, SCALE_, nullptr);

  // softmax stats, then fused normalize-in-place + P@V
  softmax_stats<<<65536, 256, 0, stream>>>(attn, stats);
  attn_pv<<<64 * 64, 32, 0, stream>>>(attn, stats, v_b, outw);
  merge_heads_cvt<<<(unsigned)((8L * 1024 * 512 + 255) / 256), blk, 0, stream>>>(outw, out_b);

  // logits = out @ Wout^T + bout
  gemm_nt_bf16<<<dim3(512 / BN, 8192 / BM, 1), 256, 0, stream>>>(
      out_b, 512, 0, wout_b, 512, 0, logits, 512, 0, 512, 1.0f, bout);

  // conv as GEMM: A = 4 flattened mem rows (K=2048 contiguous), batched over b -> new_cmem
  gemm_nt_bf16<<<dim3(512 / BN, 256 / BM, 8), 256, 0, stream>>>(
      kvin + 256L * 512, 2048, 2304L * 512, cw_b, 2048, 0,
      newcmem, 512, 256L * 512, 2048, 1.0f, convb);
  cvt_bf16<<<(unsigned)((8L * 256 * 512 + 255) / 256), blk, 0, stream>>>(newcmem, cc_b, 8L * 256 * 512);

  // ckv = compressed_mem @ Wkv^T (flat M = 2048)
  gemm_nt_bf16<<<dim3(1024 / BN, 2048 / BM, 1), 256, 0, stream>>>(
      cc_b, 512, 0, wkv_b, 512, 0, ckv_f, 1024, 0, 512, 1.0f, nullptr);
  split_heads_cvt<<<(unsigned)((8L * 8 * 256 * 64 + 255) / 256), blk, 0, stream>>>(ckv_f, ck_b, 256, 1024, 0);
  split_heads_cvt<<<(unsigned)((8L * 8 * 256 * 64 + 255) / 256), blk, 0, stream>>>(ckv_f, cv_b, 256, 1024, 512);

  // aux loss (flash, both branches), ae_loss = 0
  init_scalars<<<1, 32, 0, stream>>>(auxp, aep);
  aux_flash<<<64 * 64, 32, 0, stream>>>(q_b, k_b, v_b, ck_b, cv_b, auxp);

  // new_mem = x
  hipMemcpyAsync(newmem, x, 4UL * 8 * 1024 * 512, hipMemcpyDeviceToDevice, stream);
}
